// LightMutilHeadSelfAttention_21500606283960
// MI455X (gfx1250) — matrix operand order repaired and checked by reading, not compiled, not run on hardware
//
#include <hip/hip_runtime.h>
#include <hip/hip_bf16.h>
#include <math.h>

// ---------------- problem constants ----------------
#define CDIM    512
#define HEADS   8
#define HD      64
#define FSZ     56
#define NQ      3136           // 56*56 queries
#define KNK     784            // 28*28 keys
#define KNP     800            // keys padded to multiple of 32 (25 chunks)
#define BATCH   4
#define ATT_SCALE 0.125f       // 64^-0.5
#define LN_EPS  1e-5f
#define POSW    111            // 2*56-1

typedef __attribute__((ext_vector_type(16))) _Float16 v16h;
typedef __attribute__((ext_vector_type(8)))  _Float16 v8h;
typedef __attribute__((ext_vector_type(8)))  float    v8f;

// ---------------- WMMA fragment helpers (wave32 layouts per CDNA5 ISA 7.12.2) ----------------
__device__ __forceinline__ v8f wmma_f16(v16h a, v16h b, v8f c) {
  return __builtin_amdgcn_wmma_f32_16x16x32_f16(false, a, false, b, (short)0, c, false, false);
}

// A fragment 16x32 f16 from row-major A (lda in halfs), tile origin (row0,k0).
// lane m=row0+(lane&15); lanes>=16 hold K offsets +8 / +24.
__device__ __forceinline__ v16h frag_a(const _Float16* __restrict__ A, int lda, int row0, int k0) {
  int lane = threadIdx.x & 31;
  int m    = row0 + (lane & 15);
  int base = (lane >> 4) << 3;                    // 0 or 8
  const _Float16* p = A + (size_t)m * lda + k0 + base;
  union { v16h v; v8h h[2]; } u;
  u.h[0] = *(const v8h*)(p);                      // K = k0+base .. +7
  u.h[1] = *(const v8h*)(p + 16);                 // K = k0+16+base .. +7
  return u.v;
}

// B fragment 32x16 f16 from PRE-TRANSPOSED Bt (N x K row-major, ldb in halfs),
// tile origin (k0, n0). lane n=n0+(lane&15); lanes>=16 hold K offsets +8 / +24.
__device__ __forceinline__ v16h frag_b(const _Float16* __restrict__ Bt, int ldb, int n0, int k0) {
  int lane = threadIdx.x & 31;
  int n  = n0 + (lane & 15);
  int kb = k0 + ((lane >> 4) << 3);               // 0 or 8
  const _Float16* p = Bt + (size_t)n * ldb + kb;
  union { v16h v; v8h h[2]; } u;
  u.h[0] = *(const v8h*)(p);
  u.h[1] = *(const v8h*)(p + 16);
  return u.v;
}

// ---------------- layout/convert kernels ----------------
// x (B,C,3136) f32 -> xq (B*3136, 512) f16
__global__ __launch_bounds__(256) void k_make_xq(const float* __restrict__ x, _Float16* __restrict__ xq) {
  size_t tid = (size_t)blockIdx.x * blockDim.x + threadIdx.x;
  if (tid >= (size_t)BATCH * CDIM * NQ) return;
  int n   = (int)(tid % NQ);
  int c   = (int)((tid / NQ) % CDIM);
  int b   = (int)(tid / ((size_t)CDIM * NQ));
  xq[((size_t)b * NQ + n) * CDIM + c] = (_Float16)x[tid];
}

// W (K x N) f32 -> Wt (N x K) f16
__global__ __launch_bounds__(256) void k_make_wT(const float* __restrict__ W, _Float16* __restrict__ Wt,
                                                 int K, int N) {
  size_t tid = (size_t)blockIdx.x * blockDim.x + threadIdx.x;
  if (tid >= (size_t)K * N) return;
  int n = (int)(tid % N);
  int k = (int)(tid / N);
  Wt[(size_t)n * K + k] = (_Float16)W[tid];
}

// plain f32 -> f16 convert (Wsr: OIHW flat == already the transposed B we need)
__global__ __launch_bounds__(256) void k_cvt16(const float* __restrict__ src, _Float16* __restrict__ dst, size_t n) {
  size_t tid = (size_t)blockIdx.x * blockDim.x + threadIdx.x;
  if (tid < n) dst[tid] = (_Float16)src[tid];
}

// im2col for 2x2 stride-2 conv: A[(b*784+m), i*4+kh*2+kw] = x[b,i,2y+kh,2x+kw]
__global__ __launch_bounds__(256) void k_im2col(const float* __restrict__ x, _Float16* __restrict__ A) {
  size_t tid = (size_t)blockIdx.x * blockDim.x + threadIdx.x;
  if (tid >= (size_t)BATCH * KNK * 2048) return;
  int q   = (int)(tid & 2047);
  int row = (int)(tid >> 11);
  int i  = q >> 2, kh = (q >> 1) & 1, kw = q & 1;
  int b  = row / KNK, m = row % KNK;
  int y  = m / 28, xx = m % 28;
  A[tid] = (_Float16)x[(((size_t)b * CDIM + i) * NQ) + (2 * y + kh) * FSZ + (2 * xx + kw)];
}

// bias (3136 x 800): pos gather for m<784, -1e30 mask for padded keys
__global__ __launch_bounds__(256) void k_bias(const float* __restrict__ pos, const int* __restrict__ rel,
                                              float* __restrict__ bias) {
  size_t tid = (size_t)blockIdx.x * blockDim.x + threadIdx.x;
  if (tid >= (size_t)NQ * KNP) return;
  int m = (int)(tid % KNP);
  int n = (int)(tid / KNP);
  float v = -1.0e30f;
  if (m < KNK) {
    int i0 = rel[((size_t)n * NQ + m) * 2 + 0];
    int i1 = rel[((size_t)n * NQ + m) * 2 + 1];
    v = pos[i0 * POSW + i1];
  }
  bias[tid] = v;
}

// LayerNorm over 512 channels, one wave per row; f32 in, f16 out
__global__ __launch_bounds__(256) void k_ln(const float* __restrict__ xr, const float* __restrict__ gamma,
                                            const float* __restrict__ beta, _Float16* __restrict__ xkv) {
  int row  = (int)((blockIdx.x * blockDim.x + threadIdx.x) >> 5);
  int lane = threadIdx.x & 31;
  if (row >= BATCH * KNK) return;
  const float* p = xr + (size_t)row * CDIM;
  float v[16], s = 0.f, ss = 0.f;
#pragma unroll
  for (int i = 0; i < 16; ++i) { v[i] = p[lane + i * 32]; s += v[i]; ss += v[i] * v[i]; }
#pragma unroll
  for (int sft = 1; sft < 32; sft <<= 1) { s += __shfl_xor(s, sft, 32); ss += __shfl_xor(ss, sft, 32); }
  float mu   = s * (1.f / CDIM);
  float var  = ss * (1.f / CDIM) - mu * mu;
  float rstd = rsqrtf(var + LN_EPS);
#pragma unroll
  for (int i = 0; i < 16; ++i) {
    int c = lane + i * 32;
    xkv[(size_t)row * CDIM + c] = (_Float16)((v[i] - mu) * rstd * gamma[c] + beta[c]);
  }
}

// ---------------- WMMA GEMM kernels: each wave -> 16(M) x 64(N) tile ----------------
// Q projection: (B*N,512)x(512,512); epilogue scales by d^-1/2, writes (B,H,3136,64) f16
__global__ __launch_bounds__(256) void k_gemm_q(const _Float16* __restrict__ A, const _Float16* __restrict__ Bt,
                                                _Float16* __restrict__ qout) {
  int wave = (int)((blockIdx.x * blockDim.x + threadIdx.x) >> 5);
  if (wave >= 784 * 8) return;
  int lane = threadIdx.x & 31, lo = lane & 15, hi = lane >> 4;
  int row0 = (wave >> 3) * 16, tn = (wave & 7) * 64;
  v8f acc[4] = {};
  for (int k0 = 0; k0 < CDIM; k0 += 32) {
    __builtin_prefetch(A + (size_t)row0 * CDIM + k0 + 64, 0, 1);
    v16h a = frag_a(A, CDIM, row0, k0);
#pragma unroll
    for (int t = 0; t < 4; ++t) acc[t] = wmma_f16(a, frag_b(Bt, CDIM, tn + t * 16, k0), acc[t]);
  }
#pragma unroll
  for (int t = 0; t < 4; ++t) {
    int c = tn + t * 16 + lo, h = c >> 6, d = c & 63;
#pragma unroll
    for (int r = 0; r < 8; ++r) {
      int row = row0 + hi * 8 + r, b = row / NQ, n = row % NQ;
      qout[(((size_t)(b * HEADS + h) * NQ) + n) * HD + d] = (_Float16)(acc[t][r] * ATT_SCALE);
    }
  }
}

// SR conv as GEMM: (B*784,2048)x(2048,512) + b_sr -> xr f32
__global__ __launch_bounds__(256) void k_gemm_sr(const _Float16* __restrict__ A, const _Float16* __restrict__ Bt,
                                                 const float* __restrict__ bsr, float* __restrict__ xr) {
  int wave = (int)((blockIdx.x * blockDim.x + threadIdx.x) >> 5);
  if (wave >= 196 * 8) return;
  int lane = threadIdx.x & 31, lo = lane & 15, hi = lane >> 4;
  int row0 = (wave >> 3) * 16, tn = (wave & 7) * 64;
  v8f acc[4] = {};
  for (int k0 = 0; k0 < 2048; k0 += 32) {
    __builtin_prefetch(A + (size_t)row0 * 2048 + k0 + 64, 0, 1);
    v16h a = frag_a(A, 2048, row0, k0);
#pragma unroll
    for (int t = 0; t < 4; ++t) acc[t] = wmma_f16(a, frag_b(Bt, 2048, tn + t * 16, k0), acc[t]);
  }
#pragma unroll
  for (int t = 0; t < 4; ++t) {
    int c = tn + t * 16 + lo;
#pragma unroll
    for (int r = 0; r < 8; ++r)
      xr[(size_t)(row0 + hi * 8 + r) * CDIM + c] = acc[t][r] + bsr[c];
  }
}

// KV projection: (B*784,512)x(512,1024); col c=(d*16+h*2+e); scatter K row-major / V transposed
__global__ __launch_bounds__(256) void k_gemm_kv(const _Float16* __restrict__ A, const _Float16* __restrict__ Bt,
                                                 _Float16* __restrict__ kbuf, _Float16* __restrict__ vbuf) {
  int wave = (int)((blockIdx.x * blockDim.x + threadIdx.x) >> 5);
  if (wave >= 196 * 16) return;
  int lane = threadIdx.x & 31, lo = lane & 15, hi = lane >> 4;
  int row0 = (wave >> 4) * 16, tn = (wave & 15) * 64;
  v8f acc[4] = {};
  for (int k0 = 0; k0 < CDIM; k0 += 32) {
    v16h a = frag_a(A, CDIM, row0, k0);
#pragma unroll
    for (int t = 0; t < 4; ++t) acc[t] = wmma_f16(a, frag_b(Bt, CDIM, tn + t * 16, k0), acc[t]);
  }
#pragma unroll
  for (int t = 0; t < 4; ++t) {
    int c = tn + t * 16 + lo;
    int d = c >> 4, h = (c >> 1) & 7, e = c & 1;
#pragma unroll
    for (int r = 0; r < 8; ++r) {
      int row = row0 + hi * 8 + r, b = row / KNK, m = row % KNK;
      _Float16 val = (_Float16)acc[t][r];
      if (e == 0) kbuf[(((size_t)(b * HEADS + h)) * KNP + m) * HD + d] = val;       // K: (BH,800,64)
      else        vbuf[(((size_t)(b * HEADS + h)) * HD + d) * KNP + m] = val;       // V^T: (BH,64,800)
    }
  }
}

// Flash attention: one wave per 16-row query block; 25 chunks of 32 keys
__global__ __launch_bounds__(128) void k_attn(const _Float16* __restrict__ q, const _Float16* __restrict__ kbuf,
                                              const _Float16* __restrict__ vbuf, const float* __restrict__ bias,
                                              _Float16* __restrict__ obuf) {
  __shared__ _Float16 pl[4][16 * 32];
  int widx = threadIdx.x >> 5, lane = threadIdx.x & 31;
  int lo = lane & 15, hi = lane >> 4;
  int wave = blockIdx.x * 4 + widx;
  if (wave >= 32 * 196) return;
  int bh = wave / 196, tm = wave % 196;
  int row0 = tm * 16, b = bh >> 3, h = bh & 7;

  const _Float16* qp = q    + (size_t)bh * NQ  * HD;
  const _Float16* kp = kbuf + (size_t)bh * KNP * HD;
  const _Float16* vp = vbuf + (size_t)bh * HD  * KNP;

  v16h qa0 = frag_a(qp, HD, row0, 0);
  v16h qa1 = frag_a(qp, HD, row0, 32);

  float rmax[8], rsum[8];
#pragma unroll
  for (int r = 0; r < 8; ++r) { rmax[r] = -3.0e38f; rsum[r] = 0.f; }
  v8f o[4] = {};
  _Float16* myp = pl[widx];

  for (int m0 = 0; m0 < KNP; m0 += 32) {
    // scores: two 16x16 C tiles (K stored row-major == ready-made B fragments)
    v8f s0 = {}, s1 = {};
    s0 = wmma_f16(qa0, frag_b(kp, HD, m0,      0),  s0);
    s0 = wmma_f16(qa1, frag_b(kp, HD, m0,      32), s0);
    s1 = wmma_f16(qa0, frag_b(kp, HD, m0 + 16, 0),  s1);
    s1 = wmma_f16(qa1, frag_b(kp, HD, m0 + 16, 32), s1);
#pragma unroll
    for (int r = 0; r < 8; ++r) {
      int nrow = row0 + hi * 8 + r;
      s0[r] += bias[(size_t)nrow * KNP + m0 + lo];
      s1[r] += bias[(size_t)nrow * KNP + m0 + 16 + lo];
    }
    // online softmax (row stats live in 16-lane half-groups)
#pragma unroll
    for (int r = 0; r < 8; ++r) {
      float tmax = fmaxf(s0[r], s1[r]);
#pragma unroll
      for (int sft = 1; sft < 16; sft <<= 1) tmax = fmaxf(tmax, __shfl_xor(tmax, sft, 32));
      float nm = fmaxf(rmax[r], tmax);
      float f  = __expf(rmax[r] - nm);
      rmax[r] = nm;
      float p0 = __expf(s0[r] - nm), p1 = __expf(s1[r] - nm);
      float ps = p0 + p1;
#pragma unroll
      for (int sft = 1; sft < 16; sft <<= 1) ps += __shfl_xor(ps, sft, 32);
      rsum[r] = rsum[r] * f + ps;
      o[0][r] *= f; o[1][r] *= f; o[2][r] *= f; o[3][r] *= f;
      s0[r] = p0; s1[r] = p1;
    }
    // stage P (16x32 f16) through LDS to reshape C-layout -> A-fragment
#pragma unroll
    for (int r = 0; r < 8; ++r) {
      int rr = hi * 8 + r;
      myp[rr * 32 + lo]      = (_Float16)s0[r];
      myp[rr * 32 + 16 + lo] = (_Float16)s1[r];
    }
    asm volatile("s_wait_dscnt 0" ::: "memory");
    v16h pa;
    {
      const _Float16* p = myp + lo * 32 + hi * 8;
      union { v16h v; v8h hh[2]; } u;
      u.hh[0] = *(const v8h*)(p);
      u.hh[1] = *(const v8h*)(p + 16);
      pa = u.v;
    }
    // O += P * V  (V stored transposed == ready-made B fragments)
#pragma unroll
    for (int t = 0; t < 4; ++t) {
      const _Float16* p = vp + (size_t)(t * 16 + lo) * KNP + m0 + hi * 8;
      union { v16h v; v8h hh[2]; } u;
      u.hh[0] = *(const v8h*)(p);
      u.hh[1] = *(const v8h*)(p + 16);
      o[t] = wmma_f16(pa, u.v, o[t]);
    }
  }
  // normalize + store heads-concat (B*3136, 512) f16
#pragma unroll
  for (int t = 0; t < 4; ++t) {
    int d = t * 16 + lo;
#pragma unroll
    for (int r = 0; r < 8; ++r) {
      int nrow = row0 + hi * 8 + r;
      obuf[((size_t)b * NQ + nrow) * CDIM + h * HD + d] = (_Float16)(o[t][r] / rsum[r]);
    }
  }
}

// Output projection: (B*N,512)x(512,512) + bp -> NCHW f32
__global__ __launch_bounds__(256) void k_gemm_out(const _Float16* __restrict__ A, const _Float16* __restrict__ Bt,
                                                  const float* __restrict__ bp, float* __restrict__ out) {
  int wave = (int)((blockIdx.x * blockDim.x + threadIdx.x) >> 5);
  if (wave >= 784 * 8) return;
  int lane = threadIdx.x & 31, lo = lane & 15, hi = lane >> 4;
  int row0 = (wave >> 3) * 16, tn = (wave & 7) * 64;
  v8f acc[4] = {};
  for (int k0 = 0; k0 < CDIM; k0 += 32) {
    __builtin_prefetch(A + (size_t)row0 * CDIM + k0 + 64, 0, 1);
    v16h a = frag_a(A, CDIM, row0, k0);
#pragma unroll
    for (int t = 0; t < 4; ++t) acc[t] = wmma_f16(a, frag_b(Bt, CDIM, tn + t * 16, k0), acc[t]);
  }
#pragma unroll
  for (int t = 0; t < 4; ++t) {
    int c = tn + t * 16 + lo;
#pragma unroll
    for (int r = 0; r < 8; ++r) {
      int row = row0 + hi * 8 + r, b = row / NQ, n = row % NQ;
      out[((size_t)(b * CDIM) + c) * NQ + n] = acc[t][r] + bp[c];
    }
  }
}

// ---------------- launch ----------------
extern "C" void kernel_launch(void* const* d_in, const int* in_sizes, int n_in,
                              void* d_out, int out_size, void* d_ws, size_t ws_size,
                              hipStream_t stream) {
  const float* x     = (const float*)d_in[0];
  const float* Wq    = (const float*)d_in[1];
  const float* Wkv   = (const float*)d_in[2];
  const float* Wsr   = (const float*)d_in[3];
  const float* b_sr  = (const float*)d_in[4];
  const float* gamma = (const float*)d_in[5];
  const float* beta  = (const float*)d_in[6];
  const float* Wp    = (const float*)d_in[7];
  const float* bp    = (const float*)d_in[8];
  const float* pos   = (const float*)d_in[9];
  const int*   rel   = (const int*)d_in[10];
  float* out = (float*)d_out;

  // workspace carve-up (all offsets 256B aligned)
  char* ws = (char*)d_ws;
  size_t off = 0;
  auto take = [&](size_t bytes) { char* p = ws + off; off += (bytes + 255) & ~(size_t)255; return p; };
  _Float16* xq    = (_Float16*)take((size_t)BATCH * NQ * CDIM * 2);       // 12.85 MB
  _Float16* wqT   = (_Float16*)take((size_t)CDIM * CDIM * 2);
  _Float16* wkvT  = (_Float16*)take((size_t)CDIM * 2 * CDIM * 2);
  _Float16* wpT   = (_Float16*)take((size_t)CDIM * CDIM * 2);
  _Float16* wsr16 = (_Float16*)take((size_t)CDIM * 2048 * 2);
  _Float16* acol  = (_Float16*)take((size_t)BATCH * KNK * 2048 * 2);      // im2col
  float*    xr    = (float*)   take((size_t)BATCH * KNK * CDIM * 4);
  _Float16* xkv   = (_Float16*)take((size_t)BATCH * KNK * CDIM * 2);
  _Float16* qbuf  = (_Float16*)take((size_t)BATCH * HEADS * NQ * HD * 2);
  _Float16* kbuf  = (_Float16*)take((size_t)BATCH * HEADS * KNP * HD * 2);
  _Float16* vbuf  = (_Float16*)take((size_t)BATCH * HEADS * HD * KNP * 2);
  float*    bias  = (float*)   take((size_t)NQ * KNP * 4);
  _Float16* obuf  = (_Float16*)take((size_t)BATCH * NQ * CDIM * 2);

  // 1) converts / layout prep
  {
    size_t n = (size_t)BATCH * CDIM * NQ;
    k_make_xq<<<(unsigned)((n + 255) / 256), 256, 0, stream>>>(x, xq);
  }
  k_make_wT<<<(CDIM * CDIM + 255) / 256, 256, 0, stream>>>(Wq, wqT, CDIM, CDIM);
  k_make_wT<<<(CDIM * 2 * CDIM + 255) / 256, 256, 0, stream>>>(Wkv, wkvT, CDIM, 2 * CDIM);
  k_make_wT<<<(CDIM * CDIM + 255) / 256, 256, 0, stream>>>(Wp, wpT, CDIM, CDIM);
  k_cvt16<<<(CDIM * 2048 + 255) / 256, 256, 0, stream>>>(Wsr, wsr16, (size_t)CDIM * 2048);
  {
    size_t n = (size_t)BATCH * KNK * 2048;
    k_im2col<<<(unsigned)((n + 255) / 256), 256, 0, stream>>>(x, acol);
  }
  {
    size_t n = (size_t)NQ * KNP;
    k_bias<<<(unsigned)((n + 255) / 256), 256, 0, stream>>>(pos, rel, bias);
  }
  // zero K/V pads (padded keys masked by -1e30 bias; V pad contributes 0)
  hipMemsetAsync(kbuf, 0, (size_t)BATCH * HEADS * KNP * HD * 2, stream);
  hipMemsetAsync(vbuf, 0, (size_t)BATCH * HEADS * HD * KNP * 2, stream);

  // 2) WMMA GEMM pipeline
  k_gemm_q  <<<784, 256, 0, stream>>>(xq, wqT, qbuf);          // 6272 waves
  k_gemm_sr <<<196, 256, 0, stream>>>(acol, wsr16, b_sr, xr);  // 1568 waves
  k_ln      <<<392, 256, 0, stream>>>(xr, gamma, beta, xkv);   // wave per row
  k_gemm_kv <<<392, 256, 0, stream>>>(xkv, wkvT, kbuf, vbuf);  // 3136 waves
  k_attn    <<<1568, 128, 0, stream>>>(qbuf, kbuf, vbuf, bias, obuf);
  k_gemm_out<<<784, 256, 0, stream>>>(obuf, wpT, bp, out);
}
